// GraphEmbedding_5377299055032
// MI455X (gfx1250) — matrix operand order repaired and checked by reading, not compiled, not run on hardware
//
#include <hip/hip_runtime.h>
#include <hip/hip_bf16.h>

#define NN   50000
#define EE   600000
#define NFEAT 11
#define DD   128
#define NLAY 5

// LDS row stride in shorts: 136 -> 272 bytes/row, multiple of 16 (aligned b128
// fragment loads) and 68 dwords (rows land 4 banks apart -> good spread).
#define LSTR 136

typedef __attribute__((ext_vector_type(16))) __bf16 v16bf;
typedef __attribute__((ext_vector_type(8)))  float  v8f;

union Frag { uint4 q[2]; v16bf v; };

__device__ __forceinline__ unsigned short f2bf(float f) {
    unsigned u = __float_as_uint(f);
    u += 0x7FFFu + ((u >> 16) & 1u);          // round-to-nearest-even
    return (unsigned short)(u >> 16);
}

// ---------------- degree / norm ----------------
__global__ void k_zero(float* agg, float* deg) {
    int idx = blockIdx.x * 256 + threadIdx.x;
    if (idx < NN * DD) agg[idx] = 0.f;
    if (idx < NN)      deg[idx] = 0.f;
}

__global__ void k_deg(const int* __restrict__ col, float* __restrict__ deg) {
    int e = blockIdx.x * 256 + threadIdx.x;
    if (e < EE) unsafeAtomicAdd(&deg[col[e]], 1.0f);
}

__global__ void k_dinv(float* deg) {
    int i = blockIdx.x * 256 + threadIdx.x;
    if (i < NN) deg[i] = rsqrtf(deg[i] + 2.0f);   // deg becomes dinv in place
}

// ---------------- log-expansion: x = log(atoms+1) @ W_exp + b_exp ----------------
__global__ void k_expand(const float* __restrict__ atoms,
                         const float* __restrict__ Wexp,
                         const float* __restrict__ bexp,
                         float* __restrict__ x) {
    __shared__ float la[NFEAT];
    int i = blockIdx.x;
    int d = threadIdx.x;            // 128 threads
    if (d < NFEAT) la[d] = logf(atoms[i * NFEAT + d] + 1.0f);
    __syncthreads();
    float s = bexp[d];
#pragma unroll
    for (int k = 0; k < NFEAT; k++) s += la[k] * Wexp[k * DD + d];
    x[i * DD + d] = s;
}

// ---------------- h = x @ W  via bf16 WMMA (16x16x32, f32 accum) ----------------
__global__ __launch_bounds__(256) void k_gemm(const float* __restrict__ x,
                                              const float* __restrict__ W,
                                              float* __restrict__ h) {
    __shared__ __align__(16) unsigned short As[64 * LSTR];   // 64 rows of x, bf16
    __shared__ __align__(16) unsigned short Wt[128 * LSTR];  // W transposed [n][k], bf16

    const int tid = threadIdx.x;
    const int rowBase = blockIdx.x * 64;

    for (int idx = tid; idx < 64 * DD; idx += 256) {
        int r = idx >> 7, k = idx & 127;
        int grow = rowBase + r;
        float v = (grow < NN) ? x[grow * DD + k] : 0.f;
        As[r * LSTR + k] = f2bf(v);
    }
    for (int idx = tid; idx < DD * DD; idx += 256) {
        int k = idx >> 7, n = idx & 127;               // coalesced global read
        Wt[n * LSTR + k] = f2bf(W[idx]);               // transposed LDS write
    }
    __syncthreads();

    const int w    = tid >> 5;          // wave 0..7 -> column tile
    const int lane = tid & 31;
    const int half = lane >> 4;
    const int ln   = lane & 15;
    const int cb   = w * 16;

    v8f zero = {0.f,0.f,0.f,0.f,0.f,0.f,0.f,0.f};
    v8f acc[4];
#pragma unroll
    for (int i = 0; i < 4; i++) acc[i] = zero;

#pragma unroll
    for (int ks = 0; ks < 4; ks++) {                   // K = 128 in steps of 32
        // B fragment: lane = column (cb+ln), K(e) = ks*32 + half*16 + e
        // -> two aligned 16B runs: e=0..7 and e=8..15
        Frag bf;
        const unsigned short* bp = &Wt[(cb + ln) * LSTR + ks * 32 + half * 16];
        bf.q[0] = *(const uint4*)(bp);
        bf.q[1] = *(const uint4*)(bp + 8);

#pragma unroll
        for (int rt = 0; rt < 4; rt++) {               // 4 row tiles per wave
            // A fragment: lane row = rt*16+ln, K(e) = ks*32 + half*16 + e
            // -> two aligned 16B runs at +0 and +8
            Frag af;
            const unsigned short* ap = &As[(rt * 16 + ln) * LSTR + ks * 32 + half * 16];
            af.q[0] = *(const uint4*)(ap);
            af.q[1] = *(const uint4*)(ap + 8);

            acc[rt] = __builtin_amdgcn_wmma_f32_16x16x32_bf16(
                false, af.v, false, bf.v, (short)0, acc[rt], false, false);
        }
    }

#pragma unroll
    for (int rt = 0; rt < 4; rt++) {
#pragma unroll
        for (int r = 0; r < 8; r++) {                  // C/D: m = r + half*8, n = ln
            int grow = rowBase + rt * 16 + r + half * 8;
            if (grow < NN) h[grow * DD + cb + ln] = acc[rt][r];
        }
    }
}

// ---------------- edge gather-scale-scatter: agg[col] += h[row]*enorm ----------------
__global__ void k_edge(const int* __restrict__ row, const int* __restrict__ col,
                       const float* __restrict__ dinv,
                       const float* __restrict__ h, float* __restrict__ agg) {
    int e = blockIdx.x * 8 + (threadIdx.x >> 5);       // one wave per edge
    if (e >= EE) return;
    int lane = threadIdx.x & 31;
    int r = row[e], c = col[e];                        // wave-uniform -> scalar loads
    float wgt = dinv[r] * dinv[c];
    const float4 hv = *reinterpret_cast<const float4*>(h + r * DD + lane * 4);
    float* dst = agg + c * DD + lane * 4;
    unsafeAtomicAdd(dst + 0, hv.x * wgt);
    unsafeAtomicAdd(dst + 1, hv.y * wgt);
    unsafeAtomicAdd(dst + 2, hv.z * wgt);
    unsafeAtomicAdd(dst + 3, hv.w * wgt);
}

// ---------------- x = relu(agg + h*self_norm + b);  agg reset for next layer ----------------
__global__ void k_node(const float* __restrict__ dinv, const float* __restrict__ h,
                       float* __restrict__ agg, const float* __restrict__ bias,
                       float* __restrict__ x) {
    int idx = blockIdx.x * 256 + threadIdx.x;
    if (idx >= NN * DD) return;
    int i = idx >> 7, d = idx & 127;
    float di = dinv[i];
    float v = agg[idx] + h[idx] * (2.f * di * di) + bias[d];
    x[idx] = fmaxf(v, 0.f);
    agg[idx] = 0.f;
}

// ---------------- dense batch is identity here (100 nodes/graph, sorted) ----------------
__global__ void k_final(const float* __restrict__ x, float* __restrict__ out) {
    int idx = blockIdx.x * 256 + threadIdx.x;
    if (idx < NN * DD) out[idx] = x[idx];
    if (idx < NN)      out[NN * DD + idx] = 1.0f;      // mask: all true
}

extern "C" void kernel_launch(void* const* d_in, const int* in_sizes, int n_in,
                              void* d_out, int out_size, void* d_ws, size_t ws_size,
                              hipStream_t stream) {
    const float* atoms = (const float*)d_in[0];
    const int*   conn  = (const int*)d_in[1];
    // d_in[2] = batch: unused — layout is identity (exactly 100 nodes per graph)
    const float* Wexp  = (const float*)d_in[3];
    const float* bexp  = (const float*)d_in[4];
    const float* Ws    = (const float*)d_in[5];
    const float* bs    = (const float*)d_in[6];
    const int* row = conn;          // source
    const int* col = conn + EE;     // target

    float* x   = (float*)d_ws;      // NN*DD
    float* h   = x   + NN * DD;     // NN*DD
    float* agg = h   + NN * DD;     // NN*DD
    float* deg = agg + NN * DD;     // NN (becomes dinv)

    k_zero  <<<(NN * DD + 255) / 256, 256, 0, stream>>>(agg, deg);
    k_deg   <<<(EE + 255) / 256,      256, 0, stream>>>(col, deg);
    k_dinv  <<<(NN + 255) / 256,      256, 0, stream>>>(deg);
    k_expand<<<NN, DD, 0, stream>>>(atoms, Wexp, bexp, x);

    for (int l = 0; l < NLAY; l++) {
        k_gemm<<<(NN + 63) / 64,      256, 0, stream>>>(x, Ws + l * DD * DD, h);
        k_edge<<<(EE + 7) / 8,        256, 0, stream>>>(row, col, deg, h, agg);
        k_node<<<(NN * DD + 255) / 256, 256, 0, stream>>>(deg, h, agg, bs + l * DD, x);
    }

    k_final<<<(NN * DD + 255) / 256, 256, 0, stream>>>(x, (float*)d_out);
}
